// VectorQuantizer_80650895884340
// MI455X (gfx1250) — matrix operand order repaired and checked by reading, not compiled, not run on hardware
//
#include <hip/hip_runtime.h>
#include <hip/hip_bf16.h>

#define EMB 256
#define NCODES 8192
#define NROWS (8 * 4096)
#define LDSTRIDE (EMB + 8)

typedef __attribute__((ext_vector_type(16))) __bf16 v16bf;
typedef __attribute__((ext_vector_type(8)))  float  v8f;

union BfFrag {
  v16bf v;
  unsigned short h[16];
  uint4 q[2];
};

__device__ __forceinline__ unsigned short f2bf(float f) {
  // round-to-nearest-even f32 -> bf16
  unsigned u = __float_as_uint(f);
  u += 0x7fffu + ((u >> 16) & 1u);
  return (unsigned short)(u >> 16);
}

// Pass 1: codebook f32 -> bf16 (workspace) and per-code squared norm (f32).
// One wave per codebook row: 32 lanes x 8 elements = 256.
__global__ void vq_prep(const float* __restrict__ cb,
                        unsigned short* __restrict__ cbbf,
                        float* __restrict__ c2) {
  const int wid  = threadIdx.x >> 5;
  const int lane = threadIdx.x & 31;
  const int row  = blockIdx.x * 8 + wid;
  const float* src = cb + (size_t)row * EMB + lane * 8;
  float4 a = reinterpret_cast<const float4*>(src)[0];
  float4 b = reinterpret_cast<const float4*>(src)[1];
  float s = a.x * a.x + a.y * a.y + a.z * a.z + a.w * a.w +
            b.x * b.x + b.y * b.y + b.z * b.z + b.w * b.w;
  uint4 p;
  p.x = (unsigned)f2bf(a.x) | ((unsigned)f2bf(a.y) << 16);
  p.y = (unsigned)f2bf(a.z) | ((unsigned)f2bf(a.w) << 16);
  p.z = (unsigned)f2bf(b.x) | ((unsigned)f2bf(b.y) << 16);
  p.w = (unsigned)f2bf(b.z) | ((unsigned)f2bf(b.w) << 16);
  reinterpret_cast<uint4*>(cbbf + (size_t)row * EMB + lane * 8)[0] = p;
#pragma unroll
  for (int off = 16; off >= 1; off >>= 1) s += __shfl_xor(s, off, 32);
  if (lane == 0) c2[row] = s;
}

// Load one 16x256 z tile (rows rowBase..rowBase+15) as bf16 A-fragments.
// A layout per 32-K step: lane<16 -> K {k0..k0+15},
// lane>=16 -> K {k0+16..k0+31}.
__device__ __forceinline__ void load_a_tile(const float* __restrict__ z,
                                            int row, int half, BfFrag* frag) {
  const float* rowp = z + (size_t)row * EMB;
#pragma unroll
  for (int ks = 0; ks < 8; ++ks) {
    const int k0 = ks * 32 + half * 16;
    const int k1 = ks * 32 + half * 16 + 8;
    float4 a0 = reinterpret_cast<const float4*>(rowp + k0)[0];
    float4 a1 = reinterpret_cast<const float4*>(rowp + k0)[1];
    float4 b0 = reinterpret_cast<const float4*>(rowp + k1)[0];
    float4 b1 = reinterpret_cast<const float4*>(rowp + k1)[1];
    frag[ks].h[0]  = f2bf(a0.x); frag[ks].h[1]  = f2bf(a0.y);
    frag[ks].h[2]  = f2bf(a0.z); frag[ks].h[3]  = f2bf(a0.w);
    frag[ks].h[4]  = f2bf(a1.x); frag[ks].h[5]  = f2bf(a1.y);
    frag[ks].h[6]  = f2bf(a1.z); frag[ks].h[7]  = f2bf(a1.w);
    frag[ks].h[8]  = f2bf(b0.x); frag[ks].h[9]  = f2bf(b0.y);
    frag[ks].h[10] = f2bf(b0.z); frag[ks].h[11] = f2bf(b0.w);
    frag[ks].h[12] = f2bf(b1.x); frag[ks].h[13] = f2bf(b1.y);
    frag[ks].h[14] = f2bf(b1.z); frag[ks].h[15] = f2bf(b1.w);
  }
}

__device__ __forceinline__ void load_b_frag(BfFrag& f, const unsigned short* p) {
  const uint4* q = reinterpret_cast<const uint4*>(p);
  f.q[0] = q[0];
  f.q[1] = q[1];
}

// Pass 2: each wave owns TWO 16-row z tiles (32 rows) in bf16 register
// fragments. Codes stream through a DOUBLE-BUFFERED 32-code LDS tile filled
// by GLOBAL_LOAD_ASYNC_TO_LDS_B128: the DMA for tile nt+1 runs underneath the
// compute of tile nt (one barrier per tile). B fragments are register
// double-buffered so ds_load latency hides under the 4 WMMA chains.
__launch_bounds__(128, 1)
__global__ void vq_main(const float* __restrict__ z,
                        const float* __restrict__ codebook,
                        const unsigned short* __restrict__ cbbf,
                        const float* __restrict__ c2,
                        float* __restrict__ outq,
                        int* __restrict__ outi) {
  __shared__ __align__(16) unsigned short ldsB[2][32][LDSTRIDE];
  __shared__ int ldsIdx[4][32];

  const int tid     = threadIdx.x;
  const int wid     = tid >> 5;
  const int lane    = tid & 31;
  const int laneM   = lane & 15;
  const int half    = lane >> 4;
  const int rowBase = blockIdx.x * 128 + wid * 32;  // 4 waves x 32 rows

  BfFrag afragA[8], afragB[8];
  load_a_tile(z, rowBase + laneM,      half, afragA);
  load_a_tile(z, rowBase + 16 + laneM, half, afragB);

  float bestA[8], bestB[8];
  int   bidxA[8], bidxB[8];
#pragma unroll
  for (int r = 0; r < 8; ++r) {
    bestA[r] = 3.4e38f; bidxA[r] = 0;
    bestB[r] = 3.4e38f; bidxB[r] = 0;
  }

  // Staging: 128 threads x 128 B = 16 KB tile; 8 async b128 per thread.
  const int scode = tid >> 2;  // 0..31
  const int sseg  = tid & 3;   // 0..3 (64 ushorts each)
  const unsigned short* gstage = cbbf + (size_t)scode * EMB + sseg * 64;
  const unsigned ldsStage0 = (unsigned)(uintptr_t)(&ldsB[0][scode][sseg * 64]);
  const unsigned ldsStage1 = (unsigned)(uintptr_t)(&ldsB[1][scode][sseg * 64]);

  // Prologue: kick off DMA for tile 0 into buffer 0.
#pragma unroll
  for (int c = 0; c < 8; ++c) {
    asm volatile("global_load_async_to_lds_b128 %0, %1, off"
                 :
                 : "v"(ldsStage0 + c * 16), "v"(gstage + c * 8)
                 : "memory");
  }

  for (int nt = 0; nt < NCODES / 32; ++nt) {
    const int codeBase = nt * 32;
    const int buf = nt & 1;

    // Wait for this wave's DMA of tile nt, then one barrier: tile nt is ready
    // in LDS and everyone is done reading buffer (nt+1)&1 (used in nt-1).
    asm volatile("s_wait_asynccnt 0x0" ::: "memory");
    __syncthreads();

    if (codeBase + 32 < NCODES) {
      // Kick off DMA for tile nt+1 into the other buffer; runs under compute.
      const unsigned short* gsrc = gstage + (size_t)(codeBase + 32) * EMB;
      const unsigned dstLds = buf ? ldsStage0 : ldsStage1;
#pragma unroll
      for (int c = 0; c < 8; ++c) {
        asm volatile("global_load_async_to_lds_b128 %0, %1, off"
                     :
                     : "v"(dstLds + c * 16), "v"(gsrc + c * 8)
                     : "memory");
      }
      if (codeBase + 64 < NCODES)  // warm L2 two tiles ahead
        __builtin_prefetch(gstage + (size_t)(codeBase + 64) * EMB, 0, 1);
    }

    const float c2v0 = c2[codeBase + laneM];
    const float c2v1 = c2[codeBase + 16 + laneM];

    const unsigned short* base0 = &ldsB[buf][laneM][half * 16];
    const unsigned short* base1 = &ldsB[buf][16 + laneM][half * 16];

    v8f accA0 = {}, accA1 = {}, accB0 = {}, accB1 = {};
    // Register double-buffered B fragments: loads for ks+1 issue before the
    // WMMAs of ks, so ds latency hides under the XDL chains.
    BfFrag f0[2], f1[2];
    load_b_frag(f0[0], base0);
    load_b_frag(f1[0], base1);
#pragma unroll
    for (int ks = 0; ks < 8; ++ks) {
      const int cur = ks & 1;
      const int nxt = cur ^ 1;
      if (ks < 7) {
        load_b_frag(f0[nxt], base0 + (ks + 1) * 32);
        load_b_frag(f1[nxt], base1 + (ks + 1) * 32);
      }
      accA0 = __builtin_amdgcn_wmma_f32_16x16x32_bf16(
          false, afragA[ks].v, false, f0[cur].v, (short)0, accA0, false, false);
      accA1 = __builtin_amdgcn_wmma_f32_16x16x32_bf16(
          false, afragA[ks].v, false, f1[cur].v, (short)0, accA1, false, false);
      accB0 = __builtin_amdgcn_wmma_f32_16x16x32_bf16(
          false, afragB[ks].v, false, f0[cur].v, (short)0, accB0, false, false);
      accB1 = __builtin_amdgcn_wmma_f32_16x16x32_bf16(
          false, afragB[ks].v, false, f1[cur].v, (short)0, accB1, false, false);
    }

    const int myCode0 = codeBase + laneM;
    const int myCode1 = codeBase + 16 + laneM;
#pragma unroll
    for (int r = 0; r < 8; ++r) {
      // Merge the two column blocks first (block0 index < block1 index,
      // strict < keeps argmin-first semantics), then update running best.
      float dA0 = c2v0 - 2.0f * accA0[r];
      float dA1 = c2v1 - 2.0f * accA1[r];
      float dB0 = c2v0 - 2.0f * accB0[r];
      float dB1 = c2v1 - 2.0f * accB1[r];
      float mA = dA0; int iA = myCode0;
      if (dA1 < mA) { mA = dA1; iA = myCode1; }
      float mB = dB0; int iB = myCode0;
      if (dB1 < mB) { mB = dB1; iB = myCode1; }
      if (mA < bestA[r]) { bestA[r] = mA; bidxA[r] = iA; }
      if (mB < bestB[r]) { bestB[r] = mB; bidxB[r] = iB; }
    }
  }

  // Cross-lane argmin within each 16-lane N-group (xor bits 0..3 stay in group),
  // for both row blocks owned by this wave.
#pragma unroll
  for (int r = 0; r < 8; ++r) {
#pragma unroll
    for (int off = 1; off < 16; off <<= 1) {
      const float oA = __shfl_xor(bestA[r], off, 32);
      const int   jA = __shfl_xor(bidxA[r], off, 32);
      if (oA < bestA[r] || (oA == bestA[r] && jA < bidxA[r])) {
        bestA[r] = oA; bidxA[r] = jA;
      }
      const float oB = __shfl_xor(bestB[r], off, 32);
      const int   jB = __shfl_xor(bidxB[r], off, 32);
      if (oB < bestB[r] || (oB == bestB[r] && jB < bidxB[r])) {
        bestB[r] = oB; bidxB[r] = jB;
      }
    }
    // C layout: lanes 0-15 VGPR r -> row r; lanes 16-31 -> row r+8.
    if (laneM == r) {
      const int m = half * 8 + r;
      outi[rowBase + m]       = bidxA[r];
      ldsIdx[wid][m]          = bidxA[r];
      outi[rowBase + 16 + m]  = bidxB[r];
      ldsIdx[wid][16 + m]     = bidxB[r];
    }
  }
  __syncthreads();

  // Gather winning codebook rows in exact f32: 32 lanes x 8 floats per row.
#pragma unroll 4
  for (int m = 0; m < 32; ++m) {
    const int q = ldsIdx[wid][m];
    const float4* src = reinterpret_cast<const float4*>(codebook + (size_t)q * EMB);
    float4* dst = reinterpret_cast<float4*>(outq + (size_t)(rowBase + m) * EMB);
    dst[lane * 2]     = src[lane * 2];
    dst[lane * 2 + 1] = src[lane * 2 + 1];
  }
}

extern "C" void kernel_launch(void* const* d_in, const int* in_sizes, int n_in,
                              void* d_out, int out_size, void* d_ws, size_t ws_size,
                              hipStream_t stream) {
  (void)in_sizes; (void)n_in; (void)out_size; (void)ws_size;
  const float* z  = (const float*)d_in[0];  // (8,4096,256) f32
  const float* cb = (const float*)d_in[1];  // (8192,256)  f32

  unsigned short* cbbf = (unsigned short*)d_ws;                        // 4 MB bf16 codebook
  float* c2 = (float*)((char*)d_ws +
                       (size_t)NCODES * EMB * sizeof(unsigned short)); // 32 KB ||c||^2

  float* outq = (float*)d_out;                       // quantized, 8*4096*256 f32
  int*   outi = (int*)(outq + (size_t)NROWS * EMB);  // indices, 32768 i32

  vq_prep<<<NCODES / 8, 256, 0, stream>>>(cb, cbbf, c2);
  vq_main<<<NROWS / 128, 128, 0, stream>>>(z, cb, cbbf, c2, outq, outi);
}
